// DecoderAgent_1864015806896
// MI455X (gfx1250) — matrix operand order repaired and checked by reading, not compiled, not run on hardware
//
#include <hip/hip_runtime.h>
#include <hip/hip_bf16.h>
#include <math.h>

// ---------------- problem sizes ----------------
#define B_SZ    16
#define L_SZ    512
#define D_IN    8
#define D_MODEL 768
#define DEPTH   4
#define D_INNER 1536
#define D_STATE 16
#define D_CONV  4
#define DT_RANK 48
#define M_ROWS  (B_SZ * L_SZ)            // 8192
#define DBC_W   (DT_RANK + 2 * D_STATE)  // 80 (logical)
#define DBC_LD  128                      // padded row stride / GEMM N
#define DTK_PAD 64                       // dt GEMM K padded 48 -> 64

// ---------------- vector types ----------------
typedef __attribute__((ext_vector_type(16))) __bf16 v16bf;
typedef __attribute__((ext_vector_type(8)))  float  v8f;
typedef __attribute__((ext_vector_type(4)))  float  f32x4;

typedef unsigned short bf16_t;

__device__ __forceinline__ bf16_t f2bf(float f) {
    unsigned int u = __float_as_uint(f);
    unsigned int r = u + 0x7fffu + ((u >> 16) & 1u);   // RNE
    return (bf16_t)(r >> 16);
}
__device__ __forceinline__ float bf2f(bf16_t u) {
    return __uint_as_float((unsigned int)u << 16);
}

// Async global -> LDS copy (16B), tracked by ASYNCcnt (CDNA5 ISA).
// LDS dest address = low 32 bits of generic pointer (aperture truncation).
__device__ __forceinline__ void async_ld_b128(unsigned lds_off, const void* g) {
    asm volatile("global_load_async_to_lds_b128 %0, %1, off"
                 :: "v"(lds_off), "v"(g) : "memory");
}
__device__ __forceinline__ void wait_async0() {
    asm volatile("s_wait_asynccnt 0x0" ::: "memory");
}

// =====================================================================
// Fully-aligned GEMM: C[M,N] = A[M,K] (bf16, stride lda) * W[N,K]^T
// Requires M%64==0, N%128==0, K%32==0 (guaranteed via padded weights).
// MODE 0: plain. MODE 1: softplus(C + bias[n]).
// OUT  1: bf16 C. OUT 2: fp32 C and bf16 C2.
// 256 threads = 8 waves; wave tile 16x64; double-buffered LDS with
// global_load_async_to_lds_b128 overlapping WMMA compute.
// =====================================================================
template<int MODE, int OUT>
__global__ __launch_bounds__(256)
void gemm_bf16_wmma(const bf16_t* __restrict__ A, int lda,
                    const bf16_t* __restrict__ W,
                    const float* __restrict__ bias,
                    void* __restrict__ Cp, void* __restrict__ C2p,
                    int M, int N, int K) {
    constexpr int BM = 64, BN = 128, BK = 32, LDT = 40; // 40 = 32 + 8 pad
    __shared__ __align__(16) bf16_t sA[2][BM * LDT];
    __shared__ __align__(16) bf16_t sB[2][BN * LDT];

    const int tid  = threadIdx.x;
    const int m0   = blockIdx.y * BM;
    const int n0   = blockIdx.x * BN;
    const int wave = tid >> 5;
    const int lane = tid & 31;
    const int wm = wave & 3;         // 0..3 -> M strips of 16
    const int wn = wave >> 2;        // 0..1 -> N strips of 64
    const int laneRow  = lane & 15;
    const int laneHalf = lane >> 4;

    // staging coordinates (branchless: everything is in range)
    const int ar  = tid >> 2;                 // A row 0..63
    const int asg = (tid & 3) * 8;            // A k-seg (8 bf16 = 16B)
    const int br0 = tid >> 1;                 // B rows: 2 segs per row
    const int bsg0 = (tid & 1) * 8;           //   handled as 2 iters below

    v8f acc[4] = {};

    auto stage = [&](int k0, int buf) {
        async_ld_b128((unsigned)(size_t)&sA[buf][ar * LDT + asg],
                      &A[(size_t)(m0 + ar) * lda + k0 + asg]);
        #pragma unroll
        for (int it = 0; it < 2; ++it) {
            int idx = tid + it * 256;          // 0..511
            int r   = idx >> 2;
            int seg = (idx & 3) * 8;
            async_ld_b128((unsigned)(size_t)&sB[buf][r * LDT + seg],
                          &W[(size_t)(n0 + r) * K + k0 + seg]);
        }
    };

    const int nk = K / BK;
    stage(0, 0);

    for (int i = 0; i < nk; ++i) {
        wait_async0();
        __syncthreads();               // tile i resident for all waves;
                                       // also: all waves done reading buf (i+1)&1
        if (i + 1 < nk) stage((i + 1) * BK, (i + 1) & 1);

        const bf16_t* bA = sA[i & 1];
        const bf16_t* bB = sB[i & 1];

        union Frag { f32x4 f4[2]; v16bf v; };
        Frag fa;
        const int arow = wm * 16 + laneRow;
        const int kbA  = laneHalf * 8;      // lane<16: K 0-7/16-23, else 8-15/24-31
        fa.f4[0] = *reinterpret_cast<const f32x4*>(&bA[arow * LDT + kbA]);
        fa.f4[1] = *reinterpret_cast<const f32x4*>(&bA[arow * LDT + kbA + 16]);

        #pragma unroll
        for (int t = 0; t < 4; ++t) {
            Frag fb;
            const int bcol = wn * 64 + t * 16 + laneRow;    // N column
            const int kbB  = laneHalf * 8;                   // lane<16: K 0-7/16-23, else 8-15/24-31
            fb.f4[0] = *reinterpret_cast<const f32x4*>(&bB[bcol * LDT + kbB]);
            fb.f4[1] = *reinterpret_cast<const f32x4*>(&bB[bcol * LDT + kbB + 16]);
            acc[t] = __builtin_amdgcn_wmma_f32_16x16x32_bf16(
                false, fa.v, false, fb.v, (short)0, acc[t], false, false);
        }
    }

    // ---- store: lane laneRow = col, VGPR r -> row r + 8*laneHalf ----
    #pragma unroll
    for (int t = 0; t < 4; ++t) {
        const int n = n0 + wn * 64 + t * 16 + laneRow;
        #pragma unroll
        for (int r = 0; r < 8; ++r) {
            const int m = m0 + wm * 16 + laneHalf * 8 + r;
            float v = acc[t][r];
            if constexpr (MODE == 1) {
                v += bias[n];
                v = (v > 20.0f) ? v : __logf(1.0f + __expf(v));
            }
            const size_t o = (size_t)m * N + n;
            if constexpr (OUT == 1) {
                ((bf16_t*)Cp)[o] = f2bf(v);
            } else {
                ((float*)Cp)[o]   = v;
                ((bf16_t*)C2p)[o] = f2bf(v);
            }
        }
    }
}

// =====================================================================
// Weight fp32 -> bf16 conversion with optional zero padding of the
// (n, k) shape. Done once per launch; amortizes across 128 M-tiles.
// out[layer][n][k] (n<k n_out x k_out) = n<n_in && k<k_in ? in : 0
// =====================================================================
__global__ __launch_bounds__(256)
void k_cvt_pad(const float* __restrict__ in, bf16_t* __restrict__ out,
               int n_in, int k_in, int n_out, int k_out, int total) {
    int idx = blockIdx.x * 256 + threadIdx.x;
    if (idx >= total) return;
    int per   = n_out * k_out;
    int layer = idx / per;
    int rem   = idx - layer * per;
    int n = rem / k_out;
    int k = rem - n * k_out;
    float v = (n < n_in && k < k_in)
                  ? in[((size_t)layer * n_in + n) * k_in + k] : 0.0f;
    out[idx] = f2bf(v);
}

// =====================================================================
// Input projection: h[m,d] = obs[m,:8] . W[d,:8] + b[d]  (bf16 out)
// =====================================================================
__global__ __launch_bounds__(256)
void k_in_proj(const float* __restrict__ obs, const float* __restrict__ W,
               const float* __restrict__ bias, bf16_t* __restrict__ h) {
    int idx = blockIdx.x * 256 + threadIdx.x;
    if (idx >= M_ROWS * D_MODEL) return;
    int m = idx / D_MODEL, d = idx % D_MODEL;
    float acc = bias[d];
    #pragma unroll
    for (int i = 0; i < D_IN; ++i)
        acc += obs[(size_t)m * D_IN + i] * W[(size_t)d * D_IN + i];
    h[idx] = f2bf(acc);
}

// =====================================================================
// Depthwise causal conv (k=4) + bias + SiLU. bf16 in (x half of xz),
// fp32 accumulate, bf16 out.
// =====================================================================
__global__ __launch_bounds__(256)
void k_conv_silu(const bf16_t* __restrict__ xz, const float* __restrict__ cw,
                 const float* __restrict__ cb, bf16_t* __restrict__ x) {
    int idx = blockIdx.x * 256 + threadIdx.x;
    if (idx >= M_ROWS * D_INNER) return;
    int m = idx / D_INNER, d = idx % D_INNER;
    int b = m / L_SZ, l = m % L_SZ;
    float acc = cb[d];
    #pragma unroll
    for (int j = 0; j < D_CONV; ++j) {
        int ll = l - (D_CONV - 1) + j;
        if (ll >= 0)
            acc += bf2f(xz[(size_t)(b * L_SZ + ll) * (2 * D_INNER) + d]) * cw[d * D_CONV + j];
    }
    float sig = 1.0f / (1.0f + __expf(-acc));
    x[idx] = f2bf(acc * sig);
}

// =====================================================================
// Selective scan. One lane per (b, d). 16-state fp32 recurrence in
// registers, B_t/C_t (fp32, stride DBC_LD) broadcast via LDS.
// Fuses D*x skip and SiLU(z) gate.
// =====================================================================
__global__ __launch_bounds__(256)
void k_scan(const float* __restrict__ dbc, const bf16_t* __restrict__ dt,
            const bf16_t* __restrict__ x, const bf16_t* __restrict__ xz,
            const float* __restrict__ A_log, const float* __restrict__ Dp,
            bf16_t* __restrict__ y) {
    __shared__ float sBC[2 * D_STATE];
    const int chunks = D_INNER / 256;             // 6
    const int b = blockIdx.x / chunks;
    const int d = (blockIdx.x % chunks) * 256 + threadIdx.x;

    float Av[D_STATE], st[D_STATE];
    #pragma unroll
    for (int s = 0; s < D_STATE; ++s) {
        Av[s] = -__expf(A_log[(size_t)d * D_STATE + s]);
        st[s] = 0.0f;
    }
    const float Dv = Dp[d];

    for (int l = 0; l < L_SZ; ++l) {
        const size_t m = (size_t)b * L_SZ + l;
        if (threadIdx.x < 2 * D_STATE)
            sBC[threadIdx.x] = dbc[m * DBC_LD + DT_RANK + threadIdx.x];
        __syncthreads();

        const float dtv = bf2f(dt[m * D_INNER + d]);
        const float xv  = bf2f(x[m * D_INNER + d]);
        const float zv  = bf2f(xz[m * (2 * D_INNER) + D_INNER + d]);
        const float dtx = dtv * xv;
        float acc = 0.0f;
        #pragma unroll
        for (int s = 0; s < D_STATE; ++s) {
            float dA = __expf(dtv * Av[s]);
            st[s] = dA * st[s] + dtx * sBC[s];
            acc += st[s] * sBC[D_STATE + s];
        }
        float yy  = acc + Dv * xv;
        float sig = 1.0f / (1.0f + __expf(-zv));
        y[m * D_INNER + d] = f2bf(yy * (zv * sig));
        __syncthreads();
    }
}

// =====================================================================
// LayerNorm + policy/value heads. One block per (b,l) row. bf16 in.
// =====================================================================
__device__ __forceinline__ float block_reduce(float v, float* red, int tid) {
    red[tid] = v;
    __syncthreads();
    for (int s = 128; s > 0; s >>= 1) {
        if (tid < s) red[tid] += red[tid + s];
        __syncthreads();
    }
    float r = red[0];
    __syncthreads();
    return r;
}

__global__ __launch_bounds__(256)
void k_head(const bf16_t* __restrict__ h, const float* __restrict__ gamma,
            const float* __restrict__ beta, const float* __restrict__ pi_w,
            const float* __restrict__ pi_b, const float* __restrict__ v_w,
            const float* __restrict__ v_b, float* __restrict__ out) {
    __shared__ float red[256];
    const int row = blockIdx.x;
    const int tid = threadIdx.x;
    const bf16_t* hr = h + (size_t)row * D_MODEL;

    float v0 = bf2f(hr[tid]), v1 = bf2f(hr[tid + 256]), v2 = bf2f(hr[tid + 512]);
    float mu = block_reduce(v0 + v1 + v2, red, tid) * (1.0f / D_MODEL);
    float q0 = v0 - mu, q1 = v1 - mu, q2 = v2 - mu;
    float var = block_reduce(q0 * q0 + q1 * q1 + q2 * q2, red, tid) * (1.0f / D_MODEL);
    float rs = rsqrtf(var + 1e-5f);

    float n0 = q0 * rs * gamma[tid]       + beta[tid];
    float n1 = q1 * rs * gamma[tid + 256] + beta[tid + 256];
    float n2 = q2 * rs * gamma[tid + 512] + beta[tid + 512];

    float d0 = n0 * pi_w[tid] + n1 * pi_w[tid + 256] + n2 * pi_w[tid + 512];
    float d1 = n0 * pi_w[D_MODEL + tid] + n1 * pi_w[D_MODEL + tid + 256]
             + n2 * pi_w[D_MODEL + tid + 512];
    float dv = n0 * v_w[tid] + n1 * v_w[tid + 256] + n2 * v_w[tid + 512];

    d0 = block_reduce(d0, red, tid);
    d1 = block_reduce(d1, red, tid);
    dv = block_reduce(dv, red, tid);
    if (tid == 0) {
        out[(size_t)row * 2 + 0] = d0 + pi_b[0];
        out[(size_t)row * 2 + 1] = d1 + pi_b[1];
        out[(size_t)2 * M_ROWS + row] = dv + v_b[0];
    }
}

// =====================================================================
// Host launcher
// =====================================================================
extern "C" void kernel_launch(void* const* d_in, const int* in_sizes, int n_in,
                              void* d_out, int out_size, void* d_ws, size_t ws_size,
                              hipStream_t stream) {
    const float* obs        = (const float*)d_in[0];
    const float* in_proj_w  = (const float*)d_in[1];
    const float* in_proj_b  = (const float*)d_in[2];
    const float* lip_w      = (const float*)d_in[3];
    const float* conv_w     = (const float*)d_in[4];
    const float* conv_b     = (const float*)d_in[5];
    const float* x_proj_w   = (const float*)d_in[6];
    const float* dt_proj_w  = (const float*)d_in[7];
    const float* dt_proj_b  = (const float*)d_in[8];
    const float* A_log      = (const float*)d_in[9];
    const float* D_param    = (const float*)d_in[10];
    const float* out_proj_w = (const float*)d_in[11];
    const float* ln_gamma   = (const float*)d_in[12];
    const float* ln_beta    = (const float*)d_in[13];
    const float* pi_w       = (const float*)d_in[14];
    const float* pi_b       = (const float*)d_in[15];
    const float* v_w        = (const float*)d_in[16];
    const float* v_b        = (const float*)d_in[17];

    // ---------------- workspace layout ----------------
    char* ws = (char*)d_ws;
    size_t off = 0;
    auto carve = [&](size_t bytes) -> void* {
        void* p = ws + off;
        off += (bytes + 255) & ~(size_t)255;
        return p;
    };
    const size_t N_LIP = (size_t)DEPTH * 2 * D_INNER * D_MODEL;   // plain
    const size_t N_XP  = (size_t)DEPTH * DBC_LD * D_INNER;        // N padded 80->128
    const size_t N_DTP = (size_t)DEPTH * D_INNER * DTK_PAD;       // K padded 48->64
    const size_t N_OUT = (size_t)DEPTH * D_MODEL * D_INNER;       // plain

    bf16_t* wip_bf  = (bf16_t*)carve(N_LIP * 2);
    bf16_t* wxp_bf  = (bf16_t*)carve(N_XP * 2);
    bf16_t* wdt_bf  = (bf16_t*)carve(N_DTP * 2);
    bf16_t* wout_bf = (bf16_t*)carve(N_OUT * 2);

    bf16_t* hA     = (bf16_t*)carve((size_t)M_ROWS * D_MODEL * 2);
    bf16_t* hB     = (bf16_t*)carve((size_t)M_ROWS * D_MODEL * 2);
    bf16_t* xz     = (bf16_t*)carve((size_t)M_ROWS * 2 * D_INNER * 2);
    bf16_t* xact   = (bf16_t*)carve((size_t)M_ROWS * D_INNER * 2);
    float*  dbc    = (float*) carve((size_t)M_ROWS * DBC_LD * 4);
    bf16_t* dbc_bf = (bf16_t*)carve((size_t)M_ROWS * DBC_LD * 2);
    bf16_t* dt     = (bf16_t*)carve((size_t)M_ROWS * D_INNER * 2);
    bf16_t* ybuf   = (bf16_t*)carve((size_t)M_ROWS * D_INNER * 2);

    // ---------------- weight conversion + padding (once per launch) -----
    k_cvt_pad<<<(int)((N_LIP + 255) / 256), 256, 0, stream>>>(
        lip_w, wip_bf, DEPTH * 2 * D_INNER, D_MODEL, DEPTH * 2 * D_INNER, D_MODEL, (int)N_LIP);
    k_cvt_pad<<<(int)((N_XP + 255) / 256), 256, 0, stream>>>(
        x_proj_w, wxp_bf, DBC_W, D_INNER, DBC_LD, D_INNER, (int)N_XP);
    k_cvt_pad<<<(int)((N_DTP + 255) / 256), 256, 0, stream>>>(
        dt_proj_w, wdt_bf, DEPTH * D_INNER, DT_RANK, DEPTH * D_INNER, DTK_PAD, (int)N_DTP);
    k_cvt_pad<<<(int)((N_OUT + 255) / 256), 256, 0, stream>>>(
        out_proj_w, wout_bf, DEPTH * D_MODEL, D_INNER, DEPTH * D_MODEL, D_INNER, (int)N_OUT);

    k_in_proj<<<(M_ROWS * D_MODEL + 255) / 256, 256, 0, stream>>>(
        obs, in_proj_w, in_proj_b, hA);

    bf16_t* hcur = hA;
    bf16_t* hnxt = hB;
    for (int i = 0; i < DEPTH; ++i) {
        const bf16_t* wip  = wip_bf  + (size_t)i * 2 * D_INNER * D_MODEL;
        const float*  cw   = conv_w  + (size_t)i * D_INNER * D_CONV;
        const float*  cb   = conv_b  + (size_t)i * D_INNER;
        const bf16_t* wxp  = wxp_bf  + (size_t)i * DBC_LD * D_INNER;
        const bf16_t* wdt  = wdt_bf  + (size_t)i * D_INNER * DTK_PAD;
        const float*  bdt  = dt_proj_b + (size_t)i * D_INNER;
        const float*  Al   = A_log   + (size_t)i * D_INNER * D_STATE;
        const float*  Dp   = D_param + (size_t)i * D_INNER;
        const bf16_t* wout = wout_bf + (size_t)i * D_MODEL * D_INNER;

        // xz = h @ wip^T   (8192 x 3072 x 768) -> bf16
        dim3 g1(2 * D_INNER / 128, M_ROWS / 64);
        gemm_bf16_wmma<0, 1><<<g1, 256, 0, stream>>>(
            hcur, D_MODEL, wip, nullptr, xz, nullptr, M_ROWS, 2 * D_INNER, D_MODEL);

        // depthwise conv + SiLU -> bf16
        k_conv_silu<<<(M_ROWS * D_INNER + 255) / 256, 256, 0, stream>>>(
            xz, cw, cb, xact);

        // dbc = x @ wxp_pad^T  (8192 x 128 x 1536) -> fp32 + bf16
        dim3 g2(1, M_ROWS / 64);
        gemm_bf16_wmma<0, 2><<<g2, 256, 0, stream>>>(
            xact, D_INNER, wxp, nullptr, dbc, dbc_bf, M_ROWS, DBC_LD, D_INNER);

        // dt = softplus(dbc[:, :64] @ wdt_pad^T + bdt)  (8192 x 1536 x 64) -> bf16
        dim3 g3(D_INNER / 128, M_ROWS / 64);
        gemm_bf16_wmma<1, 1><<<g3, 256, 0, stream>>>(
            dbc_bf, DBC_LD, wdt, bdt, dt, nullptr, M_ROWS, D_INNER, DTK_PAD);

        // selective scan + gate -> bf16
        k_scan<<<B_SZ * (D_INNER / 256), 256, 0, stream>>>(
            dbc, dt, xact, xz, Al, Dp, ybuf);

        // h_next = y @ wout^T  (8192 x 768 x 1536) -> bf16
        dim3 g4(D_MODEL / 128, M_ROWS / 64);
        gemm_bf16_wmma<0, 1><<<g4, 256, 0, stream>>>(
            ybuf, D_INNER, wout, nullptr, hnxt, nullptr, M_ROWS, D_MODEL, D_INNER);

        bf16_t* tmp = hcur; hcur = hnxt; hnxt = tmp;
    }

    k_head<<<M_ROWS, 256, 0, stream>>>(
        hcur, ln_gamma, ln_beta, pi_w, pi_b, v_w, v_b, (float*)d_out);
}
